// PatchFlow_75763223101757
// MI455X (gfx1250) — matrix operand order repaired and checked by reading, not compiled, not run on hardware
//
#include <hip/hip_runtime.h>

// Problem constants (from reference)
#define Bn    32      // batch
#define Cc    3       // channels
#define HW    128     // input H=W
#define Pn    4096    // patches (64*64)
#define CONDN 16
#define HIDN  128
#define WSTR  40      // transposed weight slab row stride (halves): 80B = 16B-aligned, bank-conflict-free

typedef _Float16 v16h __attribute__((ext_vector_type(16)));
typedef _Float16 v8h  __attribute__((ext_vector_type(8)));
typedef __fp16   v2hf __attribute__((ext_vector_type(2)));   // matches __builtin_amdgcn_cvt_pkrtz return
typedef float    v8f  __attribute__((ext_vector_type(8)));

#define LOG2E 1.4426950408889634f

// Branch-free fast math on hardware transcendentals (v_exp_f32 / v_log_f32 / v_rcp_f32)
__device__ __forceinline__ float fexp(float x)  { return __builtin_amdgcn_exp2f(x * LOG2E); }
__device__ __forceinline__ float flog(float x)  { return __builtin_amdgcn_logf(x) * 0.6931471805599453f; }
__device__ __forceinline__ float fsigmoid(float z) {
    return __builtin_amdgcn_rcpf(1.f + __builtin_amdgcn_exp2f(-z * LOG2E));
}
__device__ __forceinline__ float ftanh(float x) { return 2.f * fsigmoid(2.f * x) - 1.f; }
// tanh-form GELU: x * sigmoid(1.5957691*(x + 0.044715 x^3)) — ~6 VALU, no EXEC masking
__device__ __forceinline__ float gelu_fast(float x) {
    float t = x * (1.f + 0.044715f * x * x);
    return x * fsigmoid(1.5957691216057308f * t);
}

__device__ __forceinline__ v8f vzero8() {
    v8f z = {0.f, 0.f, 0.f, 0.f, 0.f, 0.f, 0.f, 0.f};
    return z;
}

// Stage a 32(K) x 128(N) fp32 weight slab into LDS, transposed to f16: wtr[col][k].
// Coalesced float4 global loads; k >= kmax zero-padded with a WAVE-UNIFORM branch
// (kp = i*4 + waveId), so no EXEC-mask divergence is generated.
__device__ __forceinline__ void stage_slab(const float* __restrict__ wsrc, int kmax,
                                           _Float16 (*wtr)[WSTR], int tid)
{
    #pragma unroll
    for (int i = 0; i < 4; ++i) {
        int t  = i * 128 + tid;
        int kp = t >> 5;          // 0..15  (uniform per wave)
        int cg = t & 31;          // 0..31  column group (4 cols)
        int k0 = 2 * kp;
        float4 a0, a1;
        if (k0 < kmax)
            a0 = ((const float4*)(wsrc + (size_t)k0 * HIDN))[cg];
        else
            a0 = make_float4(0.f, 0.f, 0.f, 0.f);
        if (k0 + 1 < kmax)
            a1 = ((const float4*)(wsrc + (size_t)(k0 + 1) * HIDN))[cg];
        else
            a1 = make_float4(0.f, 0.f, 0.f, 0.f);
        int col = cg * 4;
        *(v2hf*)&wtr[col + 0][k0] = __builtin_amdgcn_cvt_pkrtz(a0.x, a1.x);
        *(v2hf*)&wtr[col + 1][k0] = __builtin_amdgcn_cvt_pkrtz(a0.y, a1.y);
        *(v2hf*)&wtr[col + 2][k0] = __builtin_amdgcn_cvt_pkrtz(a0.z, a1.z);
        *(v2hf*)&wtr[col + 3][k0] = __builtin_amdgcn_cvt_pkrtz(a0.w, a1.w);
    }
}

__global__ __launch_bounds__(128) void patchflow_kernel(
    const float* __restrict__ x,   const float* __restrict__ pos,
    const float* __restrict__ w1a, const float* __restrict__ b1a,
    const float* __restrict__ w2a, const float* __restrict__ b2a,
    const float* __restrict__ w3a, const float* __restrict__ b3a,
    const float* __restrict__ gsa, const float* __restrict__ goa,
    const int*   __restrict__ perma,
    const float* __restrict__ w1b, const float* __restrict__ b1b,
    const float* __restrict__ w2b, const float* __restrict__ b2b,
    const float* __restrict__ w3b, const float* __restrict__ b3b,
    const float* __restrict__ gsb, const float* __restrict__ gob,
    const int*   __restrict__ permb,
    float* __restrict__ out)
{
    __shared__ __align__(16) _Float16 wtr[HIDN][WSTR];     // transposed f16 weight slab (10 KB)
    __shared__ __align__(16) _Float16 Apad[Bn][32];        // [x1 | cond | 0] as f16
    __shared__ __align__(16) _Float16 act1[Bn][HIDN + 8];  // layer-1 output (f16)
    __shared__ __align__(16) _Float16 act2[Bn][HIDN + 8];  // layer-2 output (f16)
    __shared__ __align__(16) float    w3t[4][HIDN];        // staged w3, TRANSPOSED: [o][i]
    __shared__ float hc[Bn][3];                            // coupling state
    __shared__ float a3s[Bn][4];                           // layer-3 outputs
    __shared__ float jacc[Bn];                             // running log-det

    const int p    = blockIdx.x;
    const int tid  = threadIdx.x;
    const int lane = tid & 31;
    const int wv   = tid >> 5;       // wave id 0..3
    const int hi16 = lane >> 4;      // 0: lanes 0-15, 1: lanes 16-31
    const int l15  = lane & 15;

    // ---- 3x3 / stride-2 avg pool feeding this patch: h[b][c] ----
    if (tid < Bn * Cc) {
        int b = tid & 31, c = tid >> 5;
        int ph = p >> 6, pw = p & 63;
        int r0 = 2 * ph - 1, c0 = 2 * pw - 1;
        float s = 0.f;
        for (int dr = 0; dr < 3; ++dr) {
            int r = r0 + dr;
            if (r < 0 || r >= HW) continue;
            for (int dc = 0; dc < 3; ++dc) {
                int cc2 = c0 + dc;
                if (cc2 < 0 || cc2 >= HW) continue;
                s += x[((b * Cc + c) * HW + r) * HW + cc2];
            }
        }
        hc[b][c] = s * (1.0f / 9.0f);
    }
    if (tid < Bn) jacc[tid] = 0.f;
    __syncthreads();

    const float* W1[2]  = {w1a, w1b};
    const float* Bi1[2] = {b1a, b1b};
    const float* W2[2]  = {w2a, w2b};
    const float* Bi2[2] = {b2a, b2b};
    const float* W3[2]  = {w3a, w3b};
    const float* Bi3[2] = {b3a, b3b};
    const float* GSp[2] = {gsa, gsb};
    const float* GOp[2] = {goa, gob};
    const int*   PMp[2] = {perma, permb};

    for (int blk = 0; blk < 2; ++blk) {
        const float* w1p = W1[blk]  + (size_t)p * (1 + CONDN) * HIDN;
        const float* b1p = Bi1[blk] + (size_t)p * HIDN;
        const float* w2p = W2[blk]  + (size_t)p * HIDN * HIDN;
        const float* b2p = Bi2[blk] + (size_t)p * HIDN;
        const float* w3p = W3[blk]  + (size_t)p * HIDN * 4;
        const float* b3p = Bi3[blk] + (size_t)p * 4;
        const float* gsp = GSp[blk] + (size_t)p * 3;
        const float* gop = GOp[blk] + (size_t)p * 3;
        const int*   pmp = PMp[blk] + (size_t)p * 3;

        // Build padded A0 = [x1 | cond(16) | zeros(15)] (32x32 f16)
        for (int e = tid; e < Bn * 32; e += 128) {
            int row = e >> 5, col = e & 31;
            float v = (col == 0) ? hc[row][0]
                                 : (col <= CONDN ? pos[p * CONDN + (col - 1)] : 0.f);
            Apad[row][col] = (_Float16)v;
        }
        // Stage w3 transposed: thread tid owns row i=tid (float4 = all 4 outputs)
        {
            float4 wr = ((const float4*)w3p)[tid];
            w3t[0][tid] = wr.x; w3t[1][tid] = wr.y;
            w3t[2][tid] = wr.z; w3t[3][tid] = wr.w;
        }
        // Transposed/zero-padded w1 slab
        stage_slab(w1p, 1 + CONDN, wtr, tid);
        // prefetch first layer-2 slab while layer 1 runs
        __builtin_prefetch(w2p + (size_t)tid * 32, 0, 1);
        __syncthreads();

        // ---------------- Layer 1: [32x32(pad 17)] x [32x128] ----------------
        v8f acc[2][2];
        for (int m = 0; m < 2; ++m)
            for (int n = 0; n < 2; ++n) acc[m][n] = vzero8();
        {
            v16h aF[2];
            #pragma unroll
            for (int m = 0; m < 2; ++m) {
                const _Float16* rowp = &Apad[m * 16 + l15][hi16 * 8];
                v8h lo  = *(const v8h*)rowp;
                v8h hiv = *(const v8h*)(rowp + 16);
                aF[m] = __builtin_shufflevector(lo, hiv,
                          0,1,2,3,4,5,6,7,8,9,10,11,12,13,14,15);
            }
            #pragma unroll
            for (int nt = 0; nt < 2; ++nt) {
                int col = (wv * 2 + nt) * 16 + l15;
                v8h blo = *(const v8h*)&wtr[col][hi16 * 8];
                v8h bhi = *(const v8h*)&wtr[col][hi16 * 8 + 16];
                v16h bF = __builtin_shufflevector(blo, bhi,
                            0,1,2,3,4,5,6,7,8,9,10,11,12,13,14,15);
                #pragma unroll
                for (int m = 0; m < 2; ++m)
                    acc[m][nt] = __builtin_amdgcn_wmma_f32_16x16x32_f16(
                        false, aF[m], false, bF, (short)0, acc[m][nt], false, false);
            }
            #pragma unroll
            for (int nt = 0; nt < 2; ++nt) {
                int col = (wv * 2 + nt) * 16 + l15;
                float bv = b1p[col];
                #pragma unroll
                for (int m = 0; m < 2; ++m)
                    #pragma unroll
                    for (int r = 0; r < 8; ++r) {
                        int row = m * 16 + r + hi16 * 8;
                        act1[row][col] = (_Float16)gelu_fast(acc[m][nt][r] + bv);
                    }
            }
        }
        __syncthreads();   // act1 ready; wtr free for re-staging

        // ---------------- Layer 2: [32x128] x [128x128] ----------------
        for (int m = 0; m < 2; ++m)
            for (int n = 0; n < 2; ++n) acc[m][n] = vzero8();

        for (int kb = 0; kb < 4; ++kb) {
            const int kbase = kb * 32;
            stage_slab(w2p + (size_t)kbase * HIDN, 32, wtr, tid);
            if (kb < 3)   // pull next slab toward the caches while this one computes
                __builtin_prefetch(w2p + (size_t)(kbase + 32) * HIDN + (size_t)tid * 32, 0, 1);
            __syncthreads();

            v16h aF[2];
            #pragma unroll
            for (int m = 0; m < 2; ++m) {
                const _Float16* rowp = &act1[m * 16 + l15][kbase + hi16 * 8];
                v8h lo  = *(const v8h*)rowp;
                v8h hiv = *(const v8h*)(rowp + 16);
                aF[m] = __builtin_shufflevector(lo, hiv,
                          0,1,2,3,4,5,6,7,8,9,10,11,12,13,14,15);
            }
            #pragma unroll
            for (int nt = 0; nt < 2; ++nt) {
                int col = (wv * 2 + nt) * 16 + l15;
                v8h blo = *(const v8h*)&wtr[col][hi16 * 8];
                v8h bhi = *(const v8h*)&wtr[col][hi16 * 8 + 16];
                v16h bF = __builtin_shufflevector(blo, bhi,
                            0,1,2,3,4,5,6,7,8,9,10,11,12,13,14,15);
                #pragma unroll
                for (int m = 0; m < 2; ++m)
                    acc[m][nt] = __builtin_amdgcn_wmma_f32_16x16x32_f16(
                        false, aF[m], false, bF, (short)0, acc[m][nt], false, false);
            }
            __syncthreads();   // done reading wtr before next stage overwrites it
        }
        #pragma unroll
        for (int nt = 0; nt < 2; ++nt) {
            int col = (wv * 2 + nt) * 16 + l15;
            float bv = b2p[col];
            #pragma unroll
            for (int m = 0; m < 2; ++m)
                #pragma unroll
                for (int r = 0; r < 8; ++r) {
                    int row = m * 16 + r + hi16 * 8;
                    act2[row][col] = (_Float16)gelu_fast(acc[m][nt][r] + bv);
                }
        }
        __syncthreads();

        // ---------------- Layer 3: [32x128] x [128x4], float4 LDS dot ----------------
        {
            int b = tid >> 2, o = tid & 3;
            const float* wrow = &w3t[o][0];
            float s = 0.f;
            for (int i0 = 0; i0 < HIDN; i0 += 8) {
                v8h av = *(const v8h*)&act2[b][i0];
                float4 wa = *(const float4*)(wrow + i0);
                float4 wb = *(const float4*)(wrow + i0 + 4);
                s += (float)av[0] * wa.x + (float)av[1] * wa.y
                   + (float)av[2] * wa.z + (float)av[3] * wa.w
                   + (float)av[4] * wb.x + (float)av[5] * wb.y
                   + (float)av[6] * wb.z + (float)av[7] * wb.w;
            }
            a3s[b][o] = (s + b3p[o]) * 0.1f;
        }
        __syncthreads();

        // ---------------- Coupling + scale/offset + permutation ----------------
        if (tid < Bn) {
            int b = tid;
            float s0 = 2.f * ftanh(a3s[b][0]);
            float s1 = 2.f * ftanh(a3s[b][1]);
            float t0 = a3s[b][2], t1 = a3s[b][3];
            float ov[3];
            ov[0] = hc[b][0];
            ov[1] = hc[b][1] * fexp(s0) + t0;
            ov[2] = hc[b][2] * fexp(s1) + t1;
            float jl = s0 + s1;
            for (int c2 = 0; c2 < 3; ++c2) {
                float sc = 0.2f * flog(1.f + fexp(0.5f * gsp[c2]));   // softplus
                ov[c2] = ov[c2] * sc + gop[c2];
                jl += flog(sc);
            }
            float pv[3];
            for (int c2 = 0; c2 < 3; ++c2) pv[c2] = ov[pmp[c2]];
            hc[b][0] = pv[0]; hc[b][1] = pv[1]; hc[b][2] = pv[2];
            jacc[b] += jl;
        }
        __syncthreads();
    }

    // ---- write z [P,B,3] then j [P,B] (concatenated flat) ----
    if (tid < Bn) {
        for (int c2 = 0; c2 < 3; ++c2)
            out[((size_t)p * Bn + tid) * 3 + c2] = hc[tid][c2];
        out[(size_t)Pn * Bn * 3 + (size_t)p * Bn + tid] = jacc[tid];
    }
}

extern "C" void kernel_launch(void* const* d_in, const int* in_sizes, int n_in,
                              void* d_out, int out_size, void* d_ws, size_t ws_size,
                              hipStream_t stream) {
    // setup_inputs() dict order:
    // 0:x 1:pos 2:w1a 3:b1a 4:w2a 5:b2a 6:w3a 7:b3a 8:gsa 9:goa 10:perma
    // 11:w1b 12:b1b 13:w2b 14:b2b 15:w3b 16:b3b 17:gsb 18:gob 19:permb
    const float* x    = (const float*)d_in[0];
    const float* pos  = (const float*)d_in[1];
    const float* w1a_ = (const float*)d_in[2];
    const float* b1a_ = (const float*)d_in[3];
    const float* w2a_ = (const float*)d_in[4];
    const float* b2a_ = (const float*)d_in[5];
    const float* w3a_ = (const float*)d_in[6];
    const float* b3a_ = (const float*)d_in[7];
    const float* gsa_ = (const float*)d_in[8];
    const float* goa_ = (const float*)d_in[9];
    const int*   pma_ = (const int*)  d_in[10];
    const float* w1b_ = (const float*)d_in[11];
    const float* b1b_ = (const float*)d_in[12];
    const float* w2b_ = (const float*)d_in[13];
    const float* b2b_ = (const float*)d_in[14];
    const float* w3b_ = (const float*)d_in[15];
    const float* b3b_ = (const float*)d_in[16];
    const float* gsb_ = (const float*)d_in[17];
    const float* gob_ = (const float*)d_in[18];
    const int*   pmb_ = (const int*)  d_in[19];

    patchflow_kernel<<<Pn, 128, 0, stream>>>(
        x, pos,
        w1a_, b1a_, w2a_, b2a_, w3a_, b3a_, gsa_, goa_, pma_,
        w1b_, b1b_, w2b_, b2b_, w3b_, b3b_, gsb_, gob_, pmb_,
        (float*)d_out);
}
